// MoELayer_3530463117852
// MI455X (gfx1250) — matrix operand order repaired and checked by reading, not compiled, not run on hardware
//
#include <hip/hip_runtime.h>
#include <hip/hip_bf16.h>
#include <math.h>

// ---------------------------------------------------------------------------
// MI455X (gfx1250, wave32) implementation of the MoE transformer block.
// All large matmuls (~520 GFLOP total) go through V_WMMA_F32_16X16X32_BF16
// (fp32 accumulate). Tiles are staged into LDS as fp32 via CDNA5
// GLOBAL_LOAD_ASYNC_TO_LDS_B128 (ASYNCcnt double-buffering) when the
// toolchain exposes the builtin, else vectorized float4 staging; bf16
// conversion happens during fragment assembly (overlaps WMMA issue).
// ---------------------------------------------------------------------------

typedef __attribute__((ext_vector_type(16))) __bf16 v16bf;
typedef __attribute__((ext_vector_type(8)))  float  v8f;
typedef __attribute__((ext_vector_type(4)))  int    v4i;

#define AS1 __attribute__((address_space(1)))
#define AS3 __attribute__((address_space(3)))

#if __has_builtin(__builtin_amdgcn_global_load_async_to_lds_b128) && \
    __has_builtin(__builtin_amdgcn_s_wait_asynccnt)
#define HAVE_ASYNC 1
#define NBUF 2
#else
#define HAVE_ASYNC 0
#define NBUF 1
#endif

static constexpr int cB = 4, cS = 1024, cD = 1024, cENC = 512;
static constexpr int cDI = 2048, cDFF = 4096, cDFE = 2048;
static constexpr int cNA = 2, cNB = 6, cE = 8, cH = 16, cDH = 64;
static constexpr int NTOK = cB * cS;            // 4096

__device__ __forceinline__ float gelu_tanh(float x) {
  float x3 = x * x * x;
  float t = tanhf(0.7978845608028654f * (x + 0.044715f * x3));
  return 0.5f * x * (1.0f + t);
}
__device__ __forceinline__ float sigmoidf(float x) { return 1.0f / (1.0f + expf(-x)); }

#if HAVE_ASYNC
// Builtin signature (from clang diagnostics): param0 = int4 addrspace(1)*
// (global src), param1 = LDS dst, then imm offset / cpol. Flat LDS addresses
// carry the LDS byte offset in the low 32 bits (ISA 10.2), so the LDS
// pointer is produced by integer truncation + addrspace(3) pointer cast.
__device__ __forceinline__ void async_g2l_b128(const float* g, float* l) {
  __builtin_amdgcn_global_load_async_to_lds_b128(
      (AS1 v4i*)(uintptr_t)g,
      (AS3 v4i*)(unsigned)(uintptr_t)l, 0, 0);
}
#endif

// ---------------------------------------------------------------------------
// Generic batched WMMA GEMM:  C[z] (op)= epi( A[z](MxK) * B[z](KxN) )
//   bKmajor=1: B stored as KxN row-major (weights);  0: B stored NxK (B^T,
//   used for Q*K^T style products).
//   Epilogue: +bias[n], act(0 none,1 gelu), *rowscale[m], +addsrc[m,n],
//   accum(C += ...).  Batch offset: z1=z/zdiv, z2=z%zdiv.
// Requires M%128==0, N%64==0, K%32==0 (true for every GEMM in this graph).
// ---------------------------------------------------------------------------
struct GemmP {
  const float* A; long long sA1, sA2; int lda;
  const float* B; long long sB1, sB2; int ldb; int bKmajor;
  float*       C; long long sC1, sC2; int ldc;
  const float* bias;
  const float* rowscale; int rsStride;
  const float* addsrc;   int ldas;
  int M, N, K;
  int act;     // 0 none, 1 gelu
  int accum;   // 0 store, 1 +=
  int zdiv;    // batch index split
};

// LDS tile strides (floats). Row strides are multiples of 4 floats so every
// 16-byte chunk lands 16B-aligned for b128 stores.
#define ASTR 36     // A: [128 rows m][32 k]  pad->36
#define BKSTR 68    // B k-major: [32 rows k][64 n] pad->68
#define BNSTR 36    // B n-major: [64 rows n][32 k] pad->36

__global__ __launch_bounds__(256) void gemm_wmma_bf16(GemmP p) {
  const int tid = threadIdx.x;
  const int z   = blockIdx.z;
  const int z1  = z / p.zdiv, z2 = z % p.zdiv;
  const long long offA = (long long)z1 * p.sA1 + (long long)z2 * p.sA2;
  const long long offB = (long long)z1 * p.sB1 + (long long)z2 * p.sB2;
  const long long offC = (long long)z1 * p.sC1 + (long long)z2 * p.sC2;

  __shared__ float Asf[NBUF][128 * ASTR];   // 18 KB per buffer
  __shared__ float Bsf[NBUF][64 * BNSTR];   // 9 KB per buffer (max layout)

  const int m0 = blockIdx.y * 128;
  const int n0 = blockIdx.x * 64;
  const int wave = tid >> 5;
  const int wm = wave & 3;          // 4 waves along M (32 rows each)
  const int wn = wave >> 2;         // 2 waves along N (32 cols each)
  const int lane = tid & 31, hi = lane >> 4, ln = lane & 15;

  v8f acc[2][2] = {};
  const int nk = p.K >> 5;

  // ---- tile staging: each thread moves 4 (A) + 2 (B) 16-byte chunks -------
  auto stageA = [&](int k0, int buf) {
#pragma unroll
    for (int i = 0; i < 4; ++i) {
      int c4 = i * 256 + tid;          // chunk id; 8 chunks per 32-float row
      int m  = c4 >> 3;
      int kc = (c4 & 7) * 4;
      const float* g = p.A + offA + (long long)(m0 + m) * p.lda + (k0 + kc);
      float* l = &Asf[buf][m * ASTR + kc];
#if HAVE_ASYNC
      async_g2l_b128(g, l);
#else
      *(float4*)l = *(const float4*)g;
#endif
    }
  };
  auto stageB = [&](int k0, int buf) {
    if (p.bKmajor) {
#pragma unroll
      for (int i = 0; i < 2; ++i) {
        int c4 = i * 256 + tid;        // 16 chunks per 64-float row
        int k  = c4 >> 4;
        int nc = (c4 & 15) * 4;
        const float* g = p.B + offB + (long long)(k0 + k) * p.ldb + (n0 + nc);
        float* l = &Bsf[buf][k * BKSTR + nc];
#if HAVE_ASYNC
        async_g2l_b128(g, l);
#else
        *(float4*)l = *(const float4*)g;
#endif
      }
    } else {
#pragma unroll
      for (int i = 0; i < 2; ++i) {
        int c4 = i * 256 + tid;        // 8 chunks per 32-float row
        int n  = c4 >> 3;
        int kc = (c4 & 7) * 4;
        const float* g = p.B + offB + (long long)(n0 + n) * p.ldb + (k0 + kc);
        float* l = &Bsf[buf][n * BNSTR + kc];
#if HAVE_ASYNC
        async_g2l_b128(g, l);
#else
        *(float4*)l = *(const float4*)g;
#endif
      }
    }
  };

  // ---- fragment build (fp32 LDS -> bf16) + 4x WMMA -------------------------
  auto compute = [&](int buf) {
    // A 16x32 layout: lane = row M; lanes 0-15 K0..15, lanes 16-31 K16..31.
    v16bf aF[2], bF[2];
#pragma unroll
    for (int i = 0; i < 2; ++i) {
      const float* ab = &Asf[buf][(wm * 32 + i * 16 + ln) * ASTR];
#pragma unroll
      for (int r = 0; r < 8; ++r) {
        int kb = 2 * r + hi * 16;
        aF[i][2 * r]     = (__bf16)ab[kb];
        aF[i][2 * r + 1] = (__bf16)ab[kb + 1];
      }
    }
    // B 32x16: lane = column N; lanes 0-15 K0..15, lanes 16-31 K16..31.
    if (p.bKmajor) {
#pragma unroll
      for (int j = 0; j < 2; ++j) {
        int col = wn * 32 + j * 16 + ln;
#pragma unroll
        for (int r = 0; r < 8; ++r) {
          int kb = 2 * r + hi * 16;
          bF[j][2 * r]     = (__bf16)Bsf[buf][kb * BKSTR + col];
          bF[j][2 * r + 1] = (__bf16)Bsf[buf][(kb + 1) * BKSTR + col];
        }
      }
    } else {
#pragma unroll
      for (int j = 0; j < 2; ++j) {
        const float* bb = &Bsf[buf][(wn * 32 + j * 16 + ln) * BNSTR];
#pragma unroll
        for (int r = 0; r < 8; ++r) {
          int kb = 2 * r + hi * 16;
          bF[j][2 * r]     = (__bf16)bb[kb];
          bF[j][2 * r + 1] = (__bf16)bb[kb + 1];
        }
      }
    }
#pragma unroll
    for (int i = 0; i < 2; ++i)
#pragma unroll
      for (int j = 0; j < 2; ++j)
        acc[i][j] = __builtin_amdgcn_wmma_f32_16x16x32_bf16(
            false, aF[i], false, bF[j], (short)0, acc[i][j], false, false);
  };

#if HAVE_ASYNC
  // Double-buffered pipeline on ASYNCcnt: each wave issues 6 async ops per
  // tile; waiting for asynccnt<=6 retires the previous tile (in-order).
  stageA(0, 0); stageB(0, 0);
  for (int kt = 0; kt < nk; ++kt) {
    const int cur = kt & 1;
    if (kt + 1 < nk) {
      stageA((kt + 1) << 5, 1 - cur);
      stageB((kt + 1) << 5, 1 - cur);
      __builtin_amdgcn_s_wait_asynccnt(6);
    } else {
      __builtin_amdgcn_s_wait_asynccnt(0);
    }
    __syncthreads();
    compute(cur);
    __syncthreads();
  }
#else
  for (int kt = 0; kt < nk; ++kt) {
    const int k0 = kt << 5;
    stageA(k0, 0); stageB(k0, 0);
    if (kt + 1 < nk && tid < 16)
      __builtin_prefetch(&p.A[offA + (long long)(m0 + tid * 8) * p.lda + (k0 + 32)], 0, 1);
    __syncthreads();
    compute(0);
    __syncthreads();
  }
#endif

  // Epilogue: C/D layout — VGPR v, lanes 0-15: M=v,N=lane; lanes16-31: M=8+v.
#pragma unroll
  for (int i = 0; i < 2; ++i)
#pragma unroll
    for (int j = 0; j < 2; ++j) {
      int mB = m0 + wm * 32 + i * 16 + hi * 8;
      int n  = n0 + wn * 32 + j * 16 + ln;
      float bia = p.bias ? p.bias[n] : 0.0f;
#pragma unroll
      for (int v = 0; v < 8; ++v) {
        int m = mB + v;
        float val = acc[i][j][v] + bia;
        if (p.act == 1) val = gelu_tanh(val);
        if (p.rowscale) val *= p.rowscale[(long long)m * p.rsStride];
        if (p.addsrc)   val += p.addsrc[(long long)m * p.ldas + n];
        float* cp = p.C + offC + (long long)m * p.ldc + n;
        if (p.accum) *cp += val; else *cp = val;
      }
    }
}

// ---------------------------------------------------------------------------
// LayerNorm over last dim (D=1024), one block per token.
// ---------------------------------------------------------------------------
__global__ __launch_bounds__(256) void layernorm_kernel(
    const float* __restrict__ x, const float* __restrict__ g,
    const float* __restrict__ b, float* __restrict__ h) {
  __shared__ float r1[256], r2[256];
  const int n = blockIdx.x, tid = threadIdx.x;
  const float* xr = x + (long long)n * cD;
  float s = 0.f, ss = 0.f;
  for (int d = tid; d < cD; d += 256) { float v = xr[d]; s += v; ss += v * v; }
  r1[tid] = s; r2[tid] = ss; __syncthreads();
  for (int t = 128; t > 0; t >>= 1) {
    if (tid < t) { r1[tid] += r1[tid + t]; r2[tid] += r2[tid + t]; }
    __syncthreads();
  }
  const float mu = r1[0] * (1.0f / cD);
  const float var = r2[0] * (1.0f / cD) - mu * mu;
  const float rstd = rsqrtf(var + 1e-5f);
  for (int d = tid; d < cD; d += 256)
    h[(long long)n * cD + d] = (xr[d] - mu) * rstd * g[d] + b[d];
}

// ---------------------------------------------------------------------------
// SSM: time-invariant per-channel linear recurrence fused with SiLU gate.
// u = [z | xs] (NTOK x 4096);  h_t = a*h_{t-1} + (1-a)*xs_t; out = h_t*silu(z_t)
// ---------------------------------------------------------------------------
__global__ void ssm_scan_kernel(const float* __restrict__ u,
                                const float* __restrict__ ssm_a,
                                float* __restrict__ gated) {
  int cg = blockIdx.x * blockDim.x + threadIdx.x;   // 0 .. B*DI-1
  if (cg >= cB * cDI) return;
  int b = cg >> 11;           // /2048
  int c = cg & (cDI - 1);
  float a = sigmoidf(ssm_a[c]);
  float oma = 1.0f - a;
  float h = 0.f;
  for (int t = 0; t < cS; ++t) {
    long long base = ((long long)(b * cS + t)) * (2 * cDI);
    float zv = u[base + c];
    float xs = u[base + cDI + c];
    h = a * h + oma * xs;
    gated[((long long)(b * cS + t)) * cDI + c] = h * (zv * sigmoidf(zv));
  }
}

// ---------------------------------------------------------------------------
// Masked softmax over rows (in place). bMode 0: b=(row/rowsPer)>>4 (attn,
// mat=b*H+h); bMode 1: b=(row/rowsPer)&3 (cross-attn, mat=b).
// ---------------------------------------------------------------------------
__global__ __launch_bounds__(256) void softmax_rows_kernel(
    float* __restrict__ data, int len, int rowsPer, float scale, int causal,
    const unsigned char* __restrict__ mask, int maskLen, int bMode) {
  __shared__ float buf[1024];
  __shared__ float red[256];
  const int r = blockIdx.x, tid = threadIdx.x;
  const int mat = r / rowsPer, q = r % rowsPer;
  const int b = (bMode == 0) ? (mat >> 4) : (mat & 3);
  float* row = data + (long long)r * len;
  float mx = -3.4e38f;
  for (int j = tid; j < len; j += 256) {
    float v = row[j] * scale;
    bool ok = (!causal || j <= q);
    if (mask) ok = ok && (mask[(long long)b * maskLen + j] != 0);
    v = ok ? v : -1e9f;
    buf[j] = v;
    mx = fmaxf(mx, v);
  }
  red[tid] = mx; __syncthreads();
  for (int t = 128; t > 0; t >>= 1) { if (tid < t) red[tid] = fmaxf(red[tid], red[tid + t]); __syncthreads(); }
  mx = red[0]; __syncthreads();
  float sm = 0.f;
  for (int j = tid; j < len; j += 256) { float e = expf(buf[j] - mx); buf[j] = e; sm += e; }
  red[tid] = sm; __syncthreads();
  for (int t = 128; t > 0; t >>= 1) { if (tid < t) red[tid] += red[tid + t]; __syncthreads(); }
  const float inv = 1.0f / red[0];
  for (int j = tid; j < len; j += 256) row[j] = buf[j] * inv;
}

// x + beta_ssm*ssm_out + beta_attn*attn_out
__global__ void residual_combine_kernel(const float* __restrict__ x,
                                        const float* __restrict__ so,
                                        const float* __restrict__ ato,
                                        const float* __restrict__ bs,
                                        const float* __restrict__ ba,
                                        float* __restrict__ xa) {
  long long i = (long long)blockIdx.x * 256 + threadIdx.x;
  int d = (int)(i & (cD - 1));
  xa[i] = x[i] + bs[d] * so[i] + ba[d] * ato[i];
}

__global__ void final_add_kernel(const float* __restrict__ xa,
                                 const float* __restrict__ delta,
                                 float* __restrict__ out) {
  long long i = (long long)blockIdx.x * 256 + threadIdx.x;
  out[i] = xa[i] + delta[i];
}

// ---------------------------------------------------------------------------
// Router: logits = [x | enc_avail] @ r_W + r_b ; softmax; top-2; renorm;
// zero type-A experts when enc unavailable; renorm. One block per token.
// ---------------------------------------------------------------------------
__global__ __launch_bounds__(256) void router_kernel(
    const float* __restrict__ xa, const float* __restrict__ rW,
    const float* __restrict__ rb, const int* __restrict__ enc_avail,
    float* __restrict__ gates) {
  __shared__ float red[256 * cE];
  const int n = blockIdx.x, tid = threadIdx.x;
  const int b = n >> 10;    // /S
  const float av = (float)enc_avail[b];
  float part[cE];
#pragma unroll
  for (int e = 0; e < cE; ++e) part[e] = 0.f;
  for (int d = tid; d < cD; d += 256) {
    float xv = xa[(long long)n * cD + d];
#pragma unroll
    for (int e = 0; e < cE; ++e) part[e] += xv * rW[d * cE + e];
  }
#pragma unroll
  for (int e = 0; e < cE; ++e) red[tid * cE + e] = part[e];
  __syncthreads();
  for (int t = 128; t > 0; t >>= 1) {
    if (tid < t)
#pragma unroll
      for (int e = 0; e < cE; ++e) red[tid * cE + e] += red[(tid + t) * cE + e];
    __syncthreads();
  }
  if (tid == 0) {
    float p[cE];
    float mx = -3.4e38f;
#pragma unroll
    for (int e = 0; e < cE; ++e) {
      p[e] = red[e] + av * rW[cD * cE + e] + rb[e];
      mx = fmaxf(mx, p[e]);
    }
    float s = 0.f;
#pragma unroll
    for (int e = 0; e < cE; ++e) { p[e] = expf(p[e] - mx); s += p[e]; }
    float inv = 1.0f / s;
#pragma unroll
    for (int e = 0; e < cE; ++e) p[e] *= inv;
    int i1 = 0;
#pragma unroll
    for (int e = 1; e < cE; ++e) if (p[e] > p[i1]) i1 = e;
    int i2 = (i1 == 0) ? 1 : 0;
#pragma unroll
    for (int e = 0; e < cE; ++e) if (e != i1 && p[e] > p[i2]) i2 = e;
    float g[cE];
#pragma unroll
    for (int e = 0; e < cE; ++e) g[e] = 0.f;
    g[i1] = p[i1]; g[i2] = p[i2];
    float gs = g[i1] + g[i2];
    float invg = 1.0f / (gs + 1e-9f);
#pragma unroll
    for (int e = 0; e < cE; ++e) g[e] *= invg;
    if (av == 0.0f) { g[0] = 0.f; g[1] = 0.f; }   // type-A experts masked
    float gs2 = 0.f;
#pragma unroll
    for (int e = 0; e < cE; ++e) gs2 += g[e];
    float invg2 = 1.0f / (gs2 + 1e-6f);
#pragma unroll
    for (int e = 0; e < cE; ++e) gates[(long long)n * cE + e] = g[e] * invg2;
  }
}

// ---------------------------------------------------------------------------
// Host orchestration
// ---------------------------------------------------------------------------
extern "C" void kernel_launch(void* const* d_in, const int* in_sizes, int n_in,
                              void* d_out, int out_size, void* d_ws, size_t ws_size,
                              hipStream_t stream) {
  (void)in_sizes; (void)n_in; (void)out_size; (void)ws_size;
  const float* x         = (const float*)d_in[0];
  const float* enc       = (const float*)d_in[1];
  const unsigned char* enc_mask = (const unsigned char*)d_in[2];
  const unsigned char* pad_mask = (const unsigned char*)d_in[3];
  const int*   enc_avail = (const int*)d_in[4];
  const float* ln_g      = (const float*)d_in[5];
  const float* ln_b      = (const float*)d_in[6];
  const float* ssm_Win   = (const float*)d_in[7];
  const float* ssm_a     = (const float*)d_in[8];
  const float* ssm_Wout  = (const float*)d_in[9];
  const float* attn_Wqkv = (const float*)d_in[10];
  const float* attn_Wo   = (const float*)d_in[11];
  const float* beta_ssm  = (const float*)d_in[12];
  const float* beta_attn = (const float*)d_in[13];
  const float* se_W1     = (const float*)d_in[14];
  const float* se_b1     = (const float*)d_in[15];
  const float* se_W2     = (const float*)d_in[16];
  const float* se_b2     = (const float*)d_in[17];
  const float* r_W       = (const float*)d_in[18];
  const float* r_b       = (const float*)d_in[19];
  const float* ea_Wq     = (const float*)d_in[20];
  const float* ea_Wk     = (const float*)d_in[21];
  const float* ea_Wv     = (const float*)d_in[22];
  const float* ea_Wo     = (const float*)d_in[23];
  const float* ea_W1     = (const float*)d_in[24];
  const float* ea_W2     = (const float*)d_in[25];
  const float* eb_W1     = (const float*)d_in[26];
  const float* eb_W2     = (const float*)d_in[27];

  // workspace bump allocator (~665 MB total)
  char* ws = (char*)d_ws;
  size_t off = 0;
  auto alloc = [&](size_t elems) -> float* {
    float* p = (float*)(ws + off);
    off = (off + elems * sizeof(float) + 255) & ~(size_t)255;
    return p;
  };
  float* h_buf   = alloc((size_t)NTOK * cD);
  float* u_buf   = alloc((size_t)NTOK * 2 * cDI);
  float* gated   = alloc((size_t)NTOK * cDI);
  float* ssm_out = alloc((size_t)NTOK * cD);
  float* qkv     = alloc((size_t)NTOK * 3 * cD);
  float* scores  = alloc((size_t)cB * cH * cS * cS);
  float* ao      = alloc((size_t)NTOK * cD);
  float* attn_o  = alloc((size_t)NTOK * cD);
  float* xa      = alloc((size_t)NTOK * cD);
  float* hid     = alloc((size_t)NTOK * cDFF);
  float* delta   = alloc((size_t)NTOK * cD);
  float* gatesb  = alloc((size_t)NTOK * cE);
  float* qA      = alloc((size_t)NTOK * cD);
  float* kA      = alloc((size_t)cB * cENC * cD);
  float* vA      = alloc((size_t)cB * cENC * cD);
  float* sA      = alloc((size_t)cB * cS * cENC);
  float* ctx     = alloc((size_t)NTOK * cD);
  float* hA      = alloc((size_t)NTOK * cD);
  float* hidE    = alloc((size_t)NTOK * cDFE);

  auto gemm = [&](GemmP& g, int Z) {
    dim3 grid(g.N / 64, g.M / 128, Z);
    hipLaunchKernelGGL(gemm_wmma_bf16, grid, dim3(256), 0, stream, g);
  };
  auto mkG = [&]() { GemmP g{}; g.zdiv = 1; g.bKmajor = 1; return g; };

  // 1) LayerNorm
  hipLaunchKernelGGL(layernorm_kernel, dim3(NTOK), dim3(256), 0, stream, x, ln_g, ln_b, h_buf);

  // 2) u = h @ ssm_Win   (4096x1024 x 1024x4096)
  { GemmP g = mkG(); g.A = h_buf; g.lda = cD; g.B = ssm_Win; g.ldb = 2 * cDI;
    g.C = u_buf; g.ldc = 2 * cDI; g.M = NTOK; g.N = 2 * cDI; g.K = cD; gemm(g, 1); }

  // 3) SSM scan + SiLU gate
  hipLaunchKernelGGL(ssm_scan_kernel, dim3((cB * cDI) / 256), dim3(256), 0, stream,
                     u_buf, ssm_a, gated);

  // 4) ssm_out = gated @ ssm_Wout
  { GemmP g = mkG(); g.A = gated; g.lda = cDI; g.B = ssm_Wout; g.ldb = cD;
    g.C = ssm_out; g.ldc = cD; g.M = NTOK; g.N = cD; g.K = cDI; gemm(g, 1); }

  // 5) qkv = h @ attn_Wqkv
  { GemmP g = mkG(); g.A = h_buf; g.lda = cD; g.B = attn_Wqkv; g.ldb = 3 * cD;
    g.C = qkv; g.ldc = 3 * cD; g.M = NTOK; g.N = 3 * cD; g.K = cD; gemm(g, 1); }

  // 6) scores[b,h] = Q K^T  (batched z = b*16+h; B operand is N-major => K^T)
  { GemmP g = mkG(); g.zdiv = cH;
    g.A = qkv; g.lda = 3 * cD; g.sA1 = (long long)cS * 3 * cD; g.sA2 = cDH;
    g.B = qkv + cD; g.ldb = 3 * cD; g.bKmajor = 0; g.sB1 = (long long)cS * 3 * cD; g.sB2 = cDH;
    g.C = scores; g.ldc = cS; g.sC1 = (long long)cH * cS * cS; g.sC2 = (long long)cS * cS;
    g.M = cS; g.N = cS; g.K = cDH; gemm(g, cB * cH); }

  // 7) causal softmax (scale 1/8) with padding mask
  hipLaunchKernelGGL(softmax_rows_kernel, dim3(cB * cH * cS), dim3(256), 0, stream,
                     scores, cS, cS, 0.125f, 1, pad_mask, cS, 0);

  // 8) ao[b,s,h*64+d] = P V
  { GemmP g = mkG(); g.zdiv = cH;
    g.A = scores; g.lda = cS; g.sA1 = (long long)cH * cS * cS; g.sA2 = (long long)cS * cS;
    g.B = qkv + 2 * cD; g.ldb = 3 * cD; g.sB1 = (long long)cS * 3 * cD; g.sB2 = cDH;
    g.C = ao; g.ldc = cD; g.sC1 = (long long)cS * cD; g.sC2 = cDH;
    g.M = cS; g.N = cDH; g.K = cS; gemm(g, cB * cH); }

  // 9) attn_out = ao @ attn_Wo
  { GemmP g = mkG(); g.A = ao; g.lda = cD; g.B = attn_Wo; g.ldb = cD;
    g.C = attn_o; g.ldc = cD; g.M = NTOK; g.N = cD; g.K = cD; gemm(g, 1); }

  // 10) xa = x + beta_ssm*ssm_out + beta_attn*attn_out
  hipLaunchKernelGGL(residual_combine_kernel, dim3((NTOK * cD) / 256), dim3(256), 0, stream,
                     x, ssm_out, attn_o, beta_ssm, beta_attn, xa);

  // 11) shared expert: hid = gelu(xa @ se_W1 + b1)
  { GemmP g = mkG(); g.A = xa; g.lda = cD; g.B = se_W1; g.ldb = cDFF;
    g.C = hid; g.ldc = cDFF; g.bias = se_b1; g.act = 1;
    g.M = NTOK; g.N = cDFF; g.K = cD; gemm(g, 1); }

  // 12) delta = hid @ se_W2 + b2   (experts accumulate into this below)
  { GemmP g = mkG(); g.A = hid; g.lda = cDFF; g.B = se_W2; g.ldb = cD;
    g.C = delta; g.ldc = cD; g.bias = se_b2;
    g.M = NTOK; g.N = cD; g.K = cDFF; gemm(g, 1); }

  // 13) router -> gates (NTOK x 8)
  hipLaunchKernelGGL(router_kernel, dim3(NTOK), dim3(256), 0, stream,
                     xa, r_W, r_b, enc_avail, gatesb);

  // 14) type-A (cross-attention) experts, sequential so delta accumulation is race-free
  for (int e = 0; e < cNA; ++e) {
    const long long wDD = (long long)e * cD * cD;
    const long long wDF = (long long)e * cD * cDFE;
    const long long wFD = (long long)e * cDFE * cD;
    { GemmP g = mkG(); g.A = xa; g.lda = cD; g.B = ea_Wq + wDD; g.ldb = cD;
      g.C = qA; g.ldc = cD; g.M = NTOK; g.N = cD; g.K = cD; gemm(g, 1); }
    { GemmP g = mkG(); g.A = enc; g.lda = cD; g.B = ea_Wk + wDD; g.ldb = cD;
      g.C = kA; g.ldc = cD; g.M = cB * cENC; g.N = cD; g.K = cD; gemm(g, 1); }
    { GemmP g = mkG(); g.A = enc; g.lda = cD; g.B = ea_Wv + wDD; g.ldb = cD;
      g.C = vA; g.ldc = cD; g.M = cB * cENC; g.N = cD; g.K = cD; gemm(g, 1); }
    // sA[b] = qA[b] kA[b]^T   (z = batch)
    { GemmP g = mkG(); g.A = qA; g.lda = cD; g.sA1 = (long long)cS * cD;
      g.B = kA; g.ldb = cD; g.bKmajor = 0; g.sB1 = (long long)cENC * cD;
      g.C = sA; g.ldc = cENC; g.sC1 = (long long)cS * cENC;
      g.M = cS; g.N = cENC; g.K = cD; gemm(g, cB); }
    hipLaunchKernelGGL(softmax_rows_kernel, dim3(cB * cS), dim3(256), 0, stream,
                       sA, cENC, cS, 0.03125f, 0, enc_mask, cENC, 1);
    // ctx[b] = pA[b] vA[b]
    { GemmP g = mkG(); g.A = sA; g.lda = cENC; g.sA1 = (long long)cS * cENC;
      g.B = vA; g.ldb = cD; g.sB1 = (long long)cENC * cD;
      g.C = ctx; g.ldc = cD; g.sC1 = (long long)cS * cD;
      g.M = cS; g.N = cD; g.K = cENC; gemm(g, cB); }
    // hA = xa + ctx @ ea_Wo
    { GemmP g = mkG(); g.A = ctx; g.lda = cD; g.B = ea_Wo + wDD; g.ldb = cD;
      g.C = hA; g.ldc = cD; g.addsrc = xa; g.ldas = cD;
      g.M = NTOK; g.N = cD; g.K = cD; gemm(g, 1); }
    // hidE = gelu(hA @ ea_W1)
    { GemmP g = mkG(); g.A = hA; g.lda = cD; g.B = ea_W1 + wDF; g.ldb = cDFE;
      g.C = hidE; g.ldc = cDFE; g.act = 1;
      g.M = NTOK; g.N = cDFE; g.K = cD; gemm(g, 1); }
    // delta += gates[:,e] * (hidE @ ea_W2)
    { GemmP g = mkG(); g.A = hidE; g.lda = cDFE; g.B = ea_W2 + wFD; g.ldb = cD;
      g.C = delta; g.ldc = cD; g.accum = 1; g.rowscale = gatesb + e; g.rsStride = cE;
      g.M = NTOK; g.N = cD; g.K = cDFE; gemm(g, 1); }
  }

  // 15) type-B (FFN) experts
  for (int e = 0; e < cNB; ++e) {
    const long long wDF = (long long)e * cD * cDFE;
    const long long wFD = (long long)e * cDFE * cD;
    { GemmP g = mkG(); g.A = xa; g.lda = cD; g.B = eb_W1 + wDF; g.ldb = cDFE;
      g.C = hidE; g.ldc = cDFE; g.act = 1;
      g.M = NTOK; g.N = cDFE; g.K = cD; gemm(g, 1); }
    { GemmP g = mkG(); g.A = hidE; g.lda = cDFE; g.B = eb_W2 + wFD; g.ldb = cD;
      g.C = delta; g.ldc = cD; g.accum = 1; g.rowscale = gatesb + cNA + e; g.rsStride = cE;
      g.M = NTOK; g.N = cD; g.K = cDFE; gemm(g, 1); }
  }

  // 16) out = xa + delta
  hipLaunchKernelGGL(final_add_kernel, dim3((NTOK * cD) / 256), dim3(256), 0, stream,
                     xa, delta, (float*)d_out);
}
